// Model_47115791237202
// MI455X (gfx1250) — matrix operand order repaired and checked by reading, not compiled, not run on hardware
//
#include <hip/hip_runtime.h>
#include <hip/hip_bf16.h>
#include <math.h>

typedef __attribute__((ext_vector_type(16))) _Float16 v16h;
typedef __attribute__((ext_vector_type(8)))  _Float16 v8h;
typedef __attribute__((ext_vector_type(8)))  float    v8f;

#define GN   50000
#define GE   200000
#define GNE  250000   // E + N (self loops)
#define HID  64
#define HEADS 4
#define ODIM 32

// ---------------------------------------------------------------------------
// CAS-based float atomic max (portable, compiles to cmpswap loop)
// ---------------------------------------------------------------------------
__device__ inline void atomicMaxFloat(float* addr, float val) {
    float old = __int_as_float(__hip_atomic_load((int*)addr, __ATOMIC_RELAXED,
                                                 __HIP_MEMORY_SCOPE_AGENT));
    while (val > old) {
        int assumed = __float_as_int(old);
        int prev = atomicCAS((int*)addr, assumed, __float_as_int(val));
        if (prev == assumed) break;
        old = __int_as_float(prev);
    }
}

// ---------------------------------------------------------------------------
// Transpose + f32->f16 convert Wl/Wr [128,256] -> WT [256,128]
// ---------------------------------------------------------------------------
__global__ void convert_w_kernel(const float* __restrict__ Wl,
                                 const float* __restrict__ Wr,
                                 _Float16* __restrict__ WlT,
                                 _Float16* __restrict__ WrT) {
    int t = blockIdx.x * blockDim.x + threadIdx.x;
    if (t >= 256 * 128) return;
    int n = t >> 7, k = t & 127;
    WlT[n * 128 + k] = (_Float16)Wl[k * 256 + n];
    WrT[n * 128 + k] = (_Float16)Wr[k * 256 + n];
}

// ---------------------------------------------------------------------------
// h0 = feat @ W_in + b_in   (exact f32; tiny GEMM).  Also h = h0.
// ---------------------------------------------------------------------------
__global__ void gemm_in_kernel(const float* __restrict__ feat,
                               const float* __restrict__ W,
                               const float* __restrict__ b,
                               float* __restrict__ h0, float* __restrict__ h) {
    int t = blockIdx.x * blockDim.x + threadIdx.x;
    if (t >= GN * HID) return;
    int node = t >> 6, c = t & 63;
    const float* fr = feat + node * 32;
    float acc = b[c];
    #pragma unroll
    for (int k = 0; k < 32; ++k) acc += fr[k] * W[k * HID + c];
    h0[t] = acc;
    h[t]  = acc;
}

// ---------------------------------------------------------------------------
// Per-step: pack x=[h|h0] to f16 [N,128] and reset smax/denom/agg2
// ---------------------------------------------------------------------------
__global__ void pack_init_kernel(const float* __restrict__ h,
                                 const float* __restrict__ h0,
                                 _Float16* __restrict__ xh,
                                 float* __restrict__ agg2,
                                 float* __restrict__ smax,
                                 float* __restrict__ denom) {
    int t = blockIdx.x * blockDim.x + threadIdx.x;
    if (t >= GN * HID) return;
    int node = t >> 6, c = t & 63;
    xh[node * 128 + c]      = (_Float16)h[t];
    xh[node * 128 + 64 + c] = (_Float16)h0[t];
    agg2[t] = 0.0f;
    if (c < HEADS) {
        smax[node * HEADS + c]  = -1e30f;
        denom[node * HEADS + c] = 0.0f;
    }
}

// ---------------------------------------------------------------------------
// WMMA GEMM: xl = x @ Wl + bl ; xr = x @ Wr + br   (x f16 [N,128], out f32 [N,256])
// One wave computes one 16x16 output tile; K=128 -> 4 x v_wmma_f32_16x16x32_f16.
// A operand layout (16-bit A 16x32): lane<16 -> row=lane, K in {0-7,16-23};
// lane>=16 -> same rows, K in {8-15,24-31}.  B (32x16): lanes 0-15 K in {0-7,16-23},
// lanes 16-31 K in {8-15,24-31}, N = lane%16 (same K order as A).
// ---------------------------------------------------------------------------
__global__ __launch_bounds__(256)
void gemm_xlxr_wmma(const _Float16* __restrict__ xh,
                    const _Float16* __restrict__ WlT,
                    const _Float16* __restrict__ WrT,
                    const float* __restrict__ bl,
                    const float* __restrict__ br,
                    float* __restrict__ xl, float* __restrict__ xr) {
    int wid  = (blockIdx.x * blockDim.x + threadIdx.x) >> 5;
    int lane = threadIdx.x & 31;
    int mtile = wid >> 5;            // 3125 M-tiles of 16 rows (N=50000 exact)
    int sub   = wid & 31;            // 32 subtiles: 16 for xl, 16 for xr
    if (mtile >= GN / 16) return;
    int which = sub >> 4;            // 0 -> Wl/xl, 1 -> Wr/xr
    int ntile = (sub & 15) << 4;

    const _Float16* WT   = which ? WrT : WlT;
    const float*    bias = which ? br  : bl;
    float*          out  = which ? xr  : xl;

    int row  = mtile * 16 + (lane & 15);
    int hi   = lane >> 4;
    int ncol = ntile + (lane & 15);
    const _Float16* arow = xh + (size_t)row * 128;
    const _Float16* bcol = WT + (size_t)ncol * 128;

    v8f c = {};
    #pragma unroll
    for (int ks = 0; ks < 4; ++ks) {
        int ka = ks * 32 + hi * 8;    // A: K base for this lane half
        int kb = ks * 32 + hi * 8;    // B: K base for this lane half
        v8h a0 = *(const v8h*)(arow + ka);
        v8h a1 = *(const v8h*)(arow + ka + 16);
        v8h b0 = *(const v8h*)(bcol + kb);
        v8h b1 = *(const v8h*)(bcol + kb + 16);
        v16h a, b;
        #pragma unroll
        for (int j = 0; j < 8; ++j) {
            a[j] = a0[j]; a[j + 8] = a1[j];
            b[j] = b0[j]; b[j + 8] = b1[j];
        }
        c = __builtin_amdgcn_wmma_f32_16x16x32_f16(false, a, false, b,
                                                   (short)0, c, false, false);
    }
    float bv = bias[ncol];
    #pragma unroll
    for (int r = 0; r < 8; ++r) {
        int orow = mtile * 16 + hi * 8 + r;   // C layout: VGPR r -> M=r / M=8+r
        out[(size_t)orow * 256 + ncol] = c[r] + bv;
    }
}

// ---------------------------------------------------------------------------
// Pass A: one wave per edge.  score[e,h] = att_h . leaky(xl[src]+xr[dst]),
// masked to -1e30 if either endpoint inactive; atomic-max into smax[dst,h].
// Lane L handles elements L*8..L*8+7 -> head = L/8, 8-lane xor reduction.
// ---------------------------------------------------------------------------
__global__ __launch_bounds__(256)
void edge_score_kernel(const float* __restrict__ xl,
                       const float* __restrict__ xr,
                       const float* __restrict__ att,
                       const int* __restrict__ ei,
                       const int* __restrict__ nmask, int step,
                       float* __restrict__ score, float* __restrict__ smax) {
    int wid  = (blockIdx.x * blockDim.x + threadIdx.x) >> 5;
    int lane = threadIdx.x & 31;
    if (wid >= GNE) return;
    int s, d;
    if (wid < GE) { s = ei[wid]; d = ei[GE + wid]; } else { s = d = wid - GE; }
    bool act = (nmask[s] > step) && (nmask[d] > step);

    int h = lane >> 3;
    const float4* pl = (const float4*)(xl + (size_t)s * 256 + lane * 8);
    const float4* pr = (const float4*)(xr + (size_t)d * 256 + lane * 8);
    float4 l0 = pl[0], l1 = pl[1];
    float4 r0 = pr[0], r1 = pr[1];
    float lv[8] = {l0.x, l0.y, l0.z, l0.w, l1.x, l1.y, l1.z, l1.w};
    float rv[8] = {r0.x, r0.y, r0.z, r0.w, r1.x, r1.y, r1.z, r1.w};
    const float* ah = att + h * 64 + (lane & 7) * 8;

    float acc = 0.0f;
    #pragma unroll
    for (int j = 0; j < 8; ++j) {
        float sv = lv[j] + rv[j];
        sv = sv > 0.0f ? sv : 0.2f * sv;   // leaky_relu, slope 0.2
        acc += sv * ah[j];
    }
    acc += __shfl_xor(acc, 1, 32);
    acc += __shfl_xor(acc, 2, 32);
    acc += __shfl_xor(acc, 4, 32);
    if ((lane & 7) == 0) {
        float sc = act ? acc : -1e30f;
        score[(size_t)wid * HEADS + h] = sc;
        atomicMaxFloat(&smax[d * HEADS + h], sc);
    }
}

// ---------------------------------------------------------------------------
// Pass B: denom[dst,h] += exp(score - smax[dst,h]).  One thread per (edge,head).
// Masked edges to active dst underflow to exactly 0; inactive dst is discarded
// at update time, so no explicit mask needed here.
// ---------------------------------------------------------------------------
__global__ void edge_denom_kernel(const float* __restrict__ score,
                                  const float* __restrict__ smax,
                                  const int* __restrict__ ei,
                                  float* __restrict__ denom) {
    int t = blockIdx.x * blockDim.x + threadIdx.x;
    if (t >= GNE * HEADS) return;
    int e = t >> 2, h = t & 3;
    int d = (e < GE) ? ei[GE + e] : e - GE;
    float ex = expf(score[t] - smax[d * HEADS + h]);
    atomicAdd(&denom[d * HEADS + h], ex);
}

// ---------------------------------------------------------------------------
// Pass C: one wave per edge.  alpha_h = exp(score-smax)/max(denom,1e-16);
// m[c] = sum_h alpha_h * xl[src,h,c]  (heads summed via xor-8/16 butterfly),
// lanes 0..7 atomic-add 8 channels each into agg2[dst, 0..63].
// ---------------------------------------------------------------------------
__global__ __launch_bounds__(256)
void edge_msg_kernel(const float* __restrict__ xl,
                     const float* __restrict__ score,
                     const float* __restrict__ smax,
                     const float* __restrict__ denom,
                     const int* __restrict__ ei,
                     float* __restrict__ agg2) {
    int wid  = (blockIdx.x * blockDim.x + threadIdx.x) >> 5;
    int lane = threadIdx.x & 31;
    if (wid >= GNE) return;
    int s, d;
    if (wid < GE) { s = ei[wid]; d = ei[GE + wid]; } else { s = d = wid - GE; }

    int h = lane >> 3;
    float sc = score[(size_t)wid * HEADS + h];
    float mx = smax[d * HEADS + h];
    float dn = denom[d * HEADS + h];
    float alpha = expf(sc - mx) / fmaxf(dn, 1e-16f);

    const float4* pl = (const float4*)(xl + (size_t)s * 256 + lane * 8);
    float4 l0 = pl[0], l1 = pl[1];
    float m[8] = {alpha * l0.x, alpha * l0.y, alpha * l0.z, alpha * l0.w,
                  alpha * l1.x, alpha * l1.y, alpha * l1.z, alpha * l1.w};
    #pragma unroll
    for (int j = 0; j < 8; ++j) {
        m[j] += __shfl_xor(m[j], 8, 32);    // fold 4 head groups: channel
        m[j] += __shfl_xor(m[j], 16, 32);   // index is lane&7 invariant
    }
    if (lane < 8) {
        float* dp = agg2 + (size_t)d * 64 + lane * 8;
        #pragma unroll
        for (int j = 0; j < 8; ++j) atomicAdd(dp + j, m[j]);
    }
}

// ---------------------------------------------------------------------------
// h[n] = tanh(agg2[n] + sum_h b_conv[h]) where node active, else unchanged
// ---------------------------------------------------------------------------
__global__ void update_h_kernel(float* __restrict__ h,
                                const float* __restrict__ agg2,
                                const float* __restrict__ b_conv,
                                const int* __restrict__ nmask, int step) {
    int t = blockIdx.x * blockDim.x + threadIdx.x;
    if (t >= GN * HID) return;
    int node = t >> 6, c = t & 63;
    if (nmask[node] > step) {
        float bs = b_conv[c] + b_conv[64 + c] + b_conv[128 + c] + b_conv[192 + c];
        h[t] = tanhf(agg2[t] + bs);
    }
}

// ---------------------------------------------------------------------------
// out = (h @ Wg + bg) * (current_state > 0)   (exact f32; tiny GEMM)
// ---------------------------------------------------------------------------
__global__ void gemm_out_kernel(const float* __restrict__ h,
                                const float* __restrict__ Wg,
                                const float* __restrict__ bg,
                                const int* __restrict__ cs,
                                float* __restrict__ out) {
    int t = blockIdx.x * blockDim.x + threadIdx.x;
    if (t >= GN * ODIM) return;
    int node = t >> 5, o = t & 31;
    const float* hr = h + node * HID;
    float acc = bg[o];
    #pragma unroll 8
    for (int k = 0; k < HID; ++k) acc += hr[k] * Wg[k * ODIM + o];
    out[t] = (cs[node] > 0) ? acc : 0.0f;
}

// ---------------------------------------------------------------------------
extern "C" void kernel_launch(void* const* d_in, const int* in_sizes, int n_in,
                              void* d_out, int out_size, void* d_ws, size_t ws_size,
                              hipStream_t stream) {
    const float* feat   = (const float*)d_in[0];
    const float* W_in   = (const float*)d_in[1];
    const float* b_in   = (const float*)d_in[2];
    const float* Wl     = (const float*)d_in[3];
    const float* bl     = (const float*)d_in[4];
    const float* Wr     = (const float*)d_in[5];
    const float* br     = (const float*)d_in[6];
    const float* att    = (const float*)d_in[7];
    const float* b_conv = (const float*)d_in[8];
    const float* Wg     = (const float*)d_in[9];
    const float* bg     = (const float*)d_in[10];
    const int*   ei     = (const int*)d_in[11];
    const int*   nmask  = (const int*)d_in[12];
    const int*   cstate = (const int*)d_in[13];
    float* out = (float*)d_out;

    // workspace carve-up (256B aligned), ~160 MB total
    char* wsp = (char*)d_ws;
    auto carve = [&](size_t bytes) {
        void* p = (void*)wsp;
        wsp += (bytes + 255) & ~(size_t)255;
        return p;
    };
    float*    h     = (float*)   carve((size_t)GN * HID * 4);
    float*    h0    = (float*)   carve((size_t)GN * HID * 4);
    _Float16* xh    = (_Float16*)carve((size_t)GN * 128 * 2);
    float*    xl    = (float*)   carve((size_t)GN * 256 * 4);
    float*    xr    = (float*)   carve((size_t)GN * 256 * 4);
    _Float16* WlT   = (_Float16*)carve((size_t)256 * 128 * 2);
    _Float16* WrT   = (_Float16*)carve((size_t)256 * 128 * 2);
    float*    score = (float*)   carve((size_t)GNE * HEADS * 4);
    float*    smax  = (float*)   carve((size_t)GN * HEADS * 4);
    float*    denom = (float*)   carve((size_t)GN * HEADS * 4);
    float*    agg2  = (float*)   carve((size_t)GN * HID * 4);

    const int B = 256;
    convert_w_kernel<<<(256 * 128 + B - 1) / B, B, 0, stream>>>(Wl, Wr, WlT, WrT);
    gemm_in_kernel<<<(GN * HID + B - 1) / B, B, 0, stream>>>(feat, W_in, b_in, h0, h);

    for (int step = 0; step < 3; ++step) {
        pack_init_kernel<<<(GN * HID + B - 1) / B, B, 0, stream>>>(
            h, h0, xh, agg2, smax, denom);
        // 3125 M-tiles * 32 subtiles = 100000 waves, 8 waves/block
        gemm_xlxr_wmma<<<(GN / 16) * 32 / 8, B, 0, stream>>>(
            xh, WlT, WrT, bl, br, xl, xr);
        // one wave per edge: 250000 waves / 8 per block
        edge_score_kernel<<<GNE / 8, B, 0, stream>>>(
            xl, xr, att, ei, nmask, step, score, smax);
        edge_denom_kernel<<<(GNE * HEADS + B - 1) / B, B, 0, stream>>>(
            score, smax, ei, denom);
        edge_msg_kernel<<<GNE / 8, B, 0, stream>>>(
            xl, score, smax, denom, ei, agg2);
        update_h_kernel<<<(GN * HID + B - 1) / B, B, 0, stream>>>(
            h, agg2, b_conv, nmask, step);
    }
    gemm_out_kernel<<<(GN * ODIM + B - 1) / B, B, 0, stream>>>(h, Wg, bg, cstate, out);
}
